// FaultAttention_27547920236978
// MI455X (gfx1250) — compile-verified, matrix operand order checked by reading, not run on hardware
//
#include <hip/hip_runtime.h>
#include <hip/hip_bf16.h>

typedef __attribute__((ext_vector_type(16))) _Float16 v16h;
typedef __attribute__((ext_vector_type(8)))  float    v8f;
typedef __attribute__((ext_vector_type(4)))  float    f4;
typedef __attribute__((ext_vector_type(4)))  unsigned int u32x4;
typedef __attribute__((ext_vector_type(8)))  unsigned int u32x8;

union ABf { v16h v; f4 f[2]; };
union H2  { unsigned int u; _Float16 h[2]; };

static constexpr int B_    = 4;
static constexpr int N_    = 2048;
static constexpr int IN_D  = 64;
static constexpr int HID   = 512;
static constexpr int HEADS = 8;
static constexpr int HD    = 64;
static constexpr int C1    = 171;   // HID/3+1
static constexpr int ROWS  = B_ * N_;   // 8192

__device__ __forceinline__ v8f wmma16(v16h a, v16h b, v8f c) {
  return __builtin_amdgcn_wmma_f32_16x16x32_f16(false, a, false, b, (short)0, c, false, false);
}

// -------------------------------------------------------------------------
// f32 -> f16 convert
__global__ __launch_bounds__(256) void fa_cvt_f16(const float* __restrict__ s,
                                                  _Float16* __restrict__ d, int n) {
  int i = blockIdx.x * 256 + threadIdx.x;
  if (i < n) d[i] = (_Float16)s[i];
}

// -------------------------------------------------------------------------
// fused 3-branch conv1d + relu -> h (f32 and f16), layout [ROWS][HID]
__global__ __launch_bounds__(256) void fa_conv(const float* __restrict__ x,
                                               const float* __restrict__ w1, const float* __restrict__ b1,
                                               const float* __restrict__ w2, const float* __restrict__ b2,
                                               const float* __restrict__ w3, const float* __restrict__ b3,
                                               float* __restrict__ hf, _Float16* __restrict__ hh) {
  const int row = blockIdx.x;            // b*N + n
  const int b   = row >> 11;
  const int n   = row & (N_ - 1);
  __shared__ float xs[7 * IN_D];
  for (int t = threadIdx.x; t < 7 * IN_D; t += 256) {
    int rr = t >> 6, i = t & 63;
    int j  = n - 3 + rr;
    xs[t]  = (j >= 0 && j < N_) ? x[((size_t)b * N_ + j) * IN_D + i] : 0.f;
  }
  __syncthreads();
  for (int c = threadIdx.x; c < HID; c += 256) {
    float acc; int ks, r0; const float* wp;
    if (c < C1)            { ks = 3; r0 = 2; wp = w1 + (size_t)c * IN_D * 3;          acc = b1[c]; }
    else if (c < 2 * C1)   { ks = 5; r0 = 1; wp = w2 + (size_t)(c - C1) * IN_D * 5;   acc = b2[c - C1]; }
    else                   { ks = 7; r0 = 0; wp = w3 + (size_t)(c - 2*C1) * IN_D * 7; acc = b3[c - 2*C1]; }
    for (int i = 0; i < IN_D; ++i)
      for (int t = 0; t < ks; ++t)
        acc = fmaf(xs[(r0 + t) * IN_D + i], wp[i * ks + t], acc);
    acc = fmaxf(acc, 0.f);
    hf[(size_t)row * HID + c] = acc;
    hh[(size_t)row * HID + c] = (_Float16)acc;
  }
}

// -------------------------------------------------------------------------
// C[m,o] = sum_k A[m,k]*W[o,k] + bias[o]; A:(M,512) f16, W:(512,512) f16.
// Block = 8 waves, tile 128x64. Fragments loaded straight from global
// (two b128 per lane) using the CDNA5 16-bit A/B VGPR layouts.
__global__ __launch_bounds__(256) void fa_gemm(const _Float16* __restrict__ A,
                                               const _Float16* __restrict__ W,
                                               const float* __restrict__ bias,
                                               float* __restrict__ Cf,
                                               _Float16* __restrict__ Ch) {
  const int wave = threadIdx.x >> 5, lane = threadIdx.x & 31;
  const int m0 = blockIdx.x * 128 + wave * 16;
  const int n0 = blockIdx.y * 64;
  const int lh = lane & 15;
  const int khi = (lane >= 16) ? 8 : 0;
  const int chi = lane >> 4;

  v8f acc[4] = {};
  const _Float16* arow = A + (size_t)(m0 + lh) * HID;
  for (int kb = 0; kb < HID; kb += 32) {
    __builtin_prefetch(arow + kb + 128, 0, 1);      // global_prefetch_b8
    ABf a;
    a.f[0] = *(const f4*)(arow + kb + khi);
    a.f[1] = *(const f4*)(arow + kb + khi + 16);
#pragma unroll
    for (int nt = 0; nt < 4; ++nt) {
      const _Float16* wrow = W + (size_t)(n0 + nt * 16 + lh) * HID + kb;
      ABf bf;
      bf.f[0] = *(const f4*)(wrow + khi);
      bf.f[1] = *(const f4*)(wrow + khi + 16);
      acc[nt] = wmma16(a.v, bf.v, acc[nt]);
    }
  }
#pragma unroll
  for (int nt = 0; nt < 4; ++nt) {
    const int col = n0 + nt * 16 + lh;
    const float bv = bias ? bias[col] : 0.f;
#pragma unroll
    for (int r = 0; r < 8; ++r) {
      const int m = m0 + r + chi * 8;
      const float v = acc[nt][r] + bv;
      if (Cf) Cf[(size_t)m * HID + col] = v;
      if (Ch) Ch[(size_t)m * HID + col] = (_Float16)v;
    }
  }
}

// -------------------------------------------------------------------------
// attention: one workgroup per (b, head, 16-query tile).
// TDM stages the 16x64 Q tile into LDS; 8 waves fill a 16x2048 f32 score
// tile in LDS via WMMA; then each wave handles 2 query rows: local-window
// branch, destructive top-32 argmax (bank-conflict-free strided scan),
// softmax + dword V gather, combine, store f16 (dword pairs).
__global__ __launch_bounds__(256) void fa_attn(const _Float16* __restrict__ Qh,
                                               const _Float16* __restrict__ Kh,
                                               const _Float16* __restrict__ Vh,
                                               _Float16* __restrict__ attn_out) {
  const int bidx = blockIdx.x;
  const int qt   = bidx & 127;
  const int head = (bidx >> 7) & 7;
  const int b    = bidx >> 10;
  const int qbase = qt * 16;
  const size_t rowbase = (size_t)b * N_;

  const int wave = threadIdx.x >> 5, lane = threadIdx.x & 31;
  const int lh = lane & 15;
  const int khi = (lane >= 16) ? 8 : 0;
  const int chi = lane >> 4;

  __shared__ float sc[16][N_];                          // 128 KB score tile
  __shared__ __align__(16) _Float16 qtile[16][HD];      // 2 KB Q tile (TDM dest)

  // ---- TDM: DMA the 16x64 f16 Q tile into LDS (D# per CDNA5 ISA ch.8)
  {
    const _Float16* qg = Qh + (rowbase + qbase) * HID + head * HD;
    const unsigned long long ga = (unsigned long long)(uintptr_t)qg;
    const unsigned int lds = (unsigned int)(uintptr_t)&qtile[0][0];
    u32x4 g0;
    g0[0] = 1u;                                         // count=1, user mode
    g0[1] = lds;                                        // lds_addr
    g0[2] = (unsigned int)ga;                           // global_addr[31:0]
    g0[3] = (unsigned int)((ga >> 32) & 0x1FFFFFFull)   // global_addr[56:32]
          | (2u << 30);                                 // type=2 (image)
    u32x8 g1;
    g1[0] = 1u << 16;                                   // data_size=1 (2 bytes)
    g1[1] = (unsigned int)HD << 16;                     // tensor_dim0 = 64
    g1[2] = 16u << 16;                                  // tensor_dim1 = 16
    g1[3] = (unsigned int)HD << 16;                     // tile_dim0 = 64
    g1[4] = 16u;                                        // tile_dim1 = 16
    g1[5] = (unsigned int)HID;                          // tensor_dim0_stride = 512
    g1[6] = 0u;
    g1[7] = 0u;
    if (wave == 0) {
      asm volatile("tensor_load_to_lds %0, %1" :: "s"(g0), "s"(g1) : "memory");
      __builtin_amdgcn_s_wait_tensorcnt(0);
    }
  }
  __syncthreads();

  // Q fragments (feature blocks k=0..31, 32..63) from LDS
  const _Float16* qrow = &qtile[lh][0];
  ABf qa0, qa1;
  qa0.f[0] = *(const f4*)(qrow + khi);       qa0.f[1] = *(const f4*)(qrow + khi + 16);
  qa1.f[0] = *(const f4*)(qrow + 32 + khi);  qa1.f[1] = *(const f4*)(qrow + 32 + khi + 16);

  // scores: wave w covers keys [w*256, w*256+256)
  for (int jt = wave * 16; jt < wave * 16 + 16; ++jt) {
    const int kbase = jt * 16;
    const _Float16* krow = Kh + (rowbase + kbase + lh) * HID + head * HD;
    ABf kb0, kb1;
    kb0.f[0] = *(const f4*)(krow + khi);      kb0.f[1] = *(const f4*)(krow + khi + 16);
    kb1.f[0] = *(const f4*)(krow + 32 + khi); kb1.f[1] = *(const f4*)(krow + 32 + khi + 16);
    v8f s = {};
    s = wmma16(qa0.v, kb0.v, s);
    s = wmma16(qa1.v, kb1.v, s);
#pragma unroll
    for (int r = 0; r < 8; ++r)
      sc[r + chi * 8][kbase + lh] = s[r] * 0.125f;   // 1/sqrt(64)
  }
  __syncthreads();

  const int d0 = 2 * lane;                     // this lane's dim pair
  for (int rr = 0; rr < 2; ++rr) {
    const int m = wave + rr * 8;
    const int n = qbase + m;

    // ---- local-window branch (scores are gscores[n, n-2..n+2]; pads score 0)
    float s5[5], mx = -1e30f;
#pragma unroll
    for (int t = 0; t < 5; ++t) {
      const int j = n + t - 2;
      s5[t] = (j >= 0 && j < N_) ? sc[m][j] : 0.0f;
      mx = fmaxf(mx, s5[t]);
    }
    float se = 0.f;
#pragma unroll
    for (int t = 0; t < 5; ++t) { s5[t] = __expf(s5[t] - mx); se += s5[t]; }
    float lw = 0.f;
#pragma unroll
    for (int t = 0; t < 5; ++t) lw += s5[t] / se;    // == 1 up to rounding (as in ref)

    float lv0 = 0.f, lv1 = 0.f;
#pragma unroll
    for (int t = 0; t < 5; ++t) {
      const int j = n + t - 2;
      if (j >= 0 && j < N_) {
        H2 p; p.u = *(const unsigned int*)(Vh + (rowbase + j) * HID + head * HD + d0);
        lv0 += (float)p.h[0];
        lv1 += (float)p.h[1];
      }
    }
    const float loc0 = 0.5f * lw * lv0;
    const float loc1 = 0.5f * lw * lv1;

    // ---- top-32 by repeated wave-argmax (destructive, stride-32 scan:
    //      consecutive lanes hit consecutive LDS banks -> conflict-free)
    float selv = 0.f; int seli = 0;
    for (int round = 0; round < 32; ++round) {
      float v = -3.0e38f; int idx = 0;
      for (int j = 0; j < 64; ++j) {
        const int ii = lane + j * 32;
        const float xv = sc[m][ii];
        if (xv > v) { v = xv; idx = ii; }
      }
      for (int off = 16; off > 0; off >>= 1) {
        const float ov = __shfl_xor(v, off, 32);
        const int   oi = __shfl_xor(idx, off, 32);
        if (ov > v || (ov == v && oi < idx)) { v = ov; idx = oi; }
      }
      if ((idx & 31) == lane) sc[m][idx] = -3.0e38f;   // owner masks it out
      if (lane == round) { selv = v; seli = idx; }
      asm volatile("s_wait_dscnt 0" ::: "memory");
      __builtin_amdgcn_wave_barrier();
    }

    // ---- softmax over the 32 selected (lane t holds t-th largest)
    float mx2 = selv;
    for (int off = 16; off > 0; off >>= 1) mx2 = fmaxf(mx2, __shfl_xor(mx2, off, 32));
    float e = __expf(selv - mx2);
    float se2 = e;
    for (int off = 16; off > 0; off >>= 1) se2 += __shfl_xor(se2, off, 32);
    const float wgt = e / se2;

    // ---- gather V rows (dword loads), combine, store (dword)
    float o0 = 0.f, o1 = 0.f;
    for (int t = 0; t < 32; ++t) {
      const float wt = __shfl(wgt, t, 32);
      const int   j  = __shfl(seli, t, 32);
      H2 p; p.u = *(const unsigned int*)(Vh + (rowbase + j) * HID + head * HD + d0);
      o0 = fmaf(wt, (float)p.h[0], o0);
      o1 = fmaf(wt, (float)p.h[1], o1);
    }
    const size_t orow = (rowbase + qbase + m) * HID + head * HD;
    H2 o;
    o.h[0] = (_Float16)(loc0 + 0.5f * o0);
    o.h[1] = (_Float16)(loc1 + 0.5f * o1);
    *(unsigned int*)(attn_out + orow + d0) = o.u;
  }
}

// -------------------------------------------------------------------------
// y = proj + h ; LayerNorm over 512 -> out (f32)
__global__ __launch_bounds__(256) void fa_ln(const float* __restrict__ proj,
                                             const float* __restrict__ hres,
                                             const float* __restrict__ g,
                                             const float* __restrict__ bb,
                                             float* __restrict__ out) {
  const int row = blockIdx.x;
  const int t   = threadIdx.x;
  __shared__ float red[256];
  const size_t base = (size_t)row * HID;
  const float y0 = proj[base + t]       + hres[base + t];
  const float y1 = proj[base + t + 256] + hres[base + t + 256];

  red[t] = y0 + y1;
  __syncthreads();
  for (int s = 128; s > 0; s >>= 1) { if (t < s) red[t] += red[t + s]; __syncthreads(); }
  const float mu = red[0] * (1.f / HID);
  __syncthreads();

  const float d0 = y0 - mu, d1 = y1 - mu;
  red[t] = d0 * d0 + d1 * d1;
  __syncthreads();
  for (int s = 128; s > 0; s >>= 1) { if (t < s) red[t] += red[t + s]; __syncthreads(); }
  const float rstd = rsqrtf(red[0] * (1.f / HID) + 1e-5f);

  out[base + t]       = d0 * rstd * g[t]       + bb[t];
  out[base + t + 256] = d1 * rstd * g[t + 256] + bb[t + 256];
}

// -------------------------------------------------------------------------
extern "C" void kernel_launch(void* const* d_in, const int* in_sizes, int n_in,
                              void* d_out, int out_size, void* d_ws, size_t ws_size,
                              hipStream_t stream) {
  (void)in_sizes; (void)n_in; (void)out_size; (void)ws_size;
  const float* x   = (const float*)d_in[0];
  const float* w1  = (const float*)d_in[1];
  const float* b1  = (const float*)d_in[2];
  const float* w2  = (const float*)d_in[3];
  const float* b2  = (const float*)d_in[4];
  const float* w3  = (const float*)d_in[5];
  const float* b3  = (const float*)d_in[6];
  const float* wq  = (const float*)d_in[7];
  const float* bq  = (const float*)d_in[8];
  const float* wk  = (const float*)d_in[9];
  const float* bk  = (const float*)d_in[10];
  const float* wv  = (const float*)d_in[11];
  const float* bv  = (const float*)d_in[12];
  const float* wo  = (const float*)d_in[13];
  const float* bo  = (const float*)d_in[14];
  const float* lng = (const float*)d_in[15];
  const float* lnb = (const float*)d_in[16];
  float* out = (float*)d_out;

  char* ws = (char*)d_ws;
  size_t off = 0;
  auto alloc = [&](size_t bytes) -> void* {
    off = (off + 255) & ~(size_t)255;
    void* p = ws + off;
    off += bytes;
    return p;
  };
  float*    hf32  = (float*)   alloc((size_t)ROWS * HID * 4);
  _Float16* hf16  = (_Float16*)alloc((size_t)ROWS * HID * 2);
  _Float16* wq16  = (_Float16*)alloc((size_t)HID * HID * 2);
  _Float16* wk16  = (_Float16*)alloc((size_t)HID * HID * 2);
  _Float16* wv16  = (_Float16*)alloc((size_t)HID * HID * 2);
  _Float16* wo16  = (_Float16*)alloc((size_t)HID * HID * 2);
  _Float16* qf16  = (_Float16*)alloc((size_t)ROWS * HID * 2);
  _Float16* kf16  = (_Float16*)alloc((size_t)ROWS * HID * 2);
  _Float16* vf16  = (_Float16*)alloc((size_t)ROWS * HID * 2);
  _Float16* att16 = (_Float16*)alloc((size_t)ROWS * HID * 2);
  float*    proj  = (float*)   alloc((size_t)ROWS * HID * 4);

  const int wn = HID * HID;
  fa_cvt_f16<<<(wn + 255) / 256, 256, 0, stream>>>(wq, wq16, wn);
  fa_cvt_f16<<<(wn + 255) / 256, 256, 0, stream>>>(wk, wk16, wn);
  fa_cvt_f16<<<(wn + 255) / 256, 256, 0, stream>>>(wv, wv16, wn);
  fa_cvt_f16<<<(wn + 255) / 256, 256, 0, stream>>>(wo, wo16, wn);

  fa_conv<<<ROWS, 256, 0, stream>>>(x, w1, b1, w2, b2, w3, b3, hf32, hf16);

  dim3 ggrid(ROWS / 128, HID / 64);
  fa_gemm<<<ggrid, 256, 0, stream>>>(hf16, wq16, bq, nullptr, qf16);
  fa_gemm<<<ggrid, 256, 0, stream>>>(hf16, wk16, bk, nullptr, kf16);
  fa_gemm<<<ggrid, 256, 0, stream>>>(hf16, wv16, bv, nullptr, vf16);

  fa_attn<<<B_ * HEADS * (N_ / 16), 256, 0, stream>>>(qf16, kf16, vf16, att16);

  fa_gemm<<<ggrid, 256, 0, stream>>>(att16, wo16, bo, proj, nullptr);

  fa_ln<<<ROWS, 256, 0, stream>>>(proj, hf32, lng, lnb, out);
}
